// PointTokenizer_34608846471751
// MI455X (gfx1250) — matrix operand order repaired and checked by reading, not compiled, not run on hardware
//
#include <hip/hip_runtime.h>

#define BATCH 8
#define NPTS  16384
#define M     512
#define K     32
#define FEAT  256
#define EPS   1e-5f

typedef __attribute__((ext_vector_type(16))) _Float16 v16h;
typedef __attribute__((ext_vector_type(8)))  _Float16 v8h;
typedef __attribute__((ext_vector_type(8)))  float    v8f;
typedef __attribute__((ext_vector_type(4)))  unsigned int u32x4;
typedef __attribute__((ext_vector_type(8)))  int      i32x8;
typedef __attribute__((ext_vector_type(4)))  int      i32x4;

#if defined(__has_builtin)
#if __has_builtin(__builtin_amdgcn_tensor_load_to_lds)
#define HAVE_TDM 1
#endif
#endif

// ---------------------------------------------------------------------------
// WMMA helpers (CDNA5 16x16x32 f16, layouts per CDNA5 ISA 7.12.2)
// ---------------------------------------------------------------------------
__device__ __forceinline__ v8f wmma16(v16h a, v16h b, v8f c) {
  return __builtin_amdgcn_wmma_f32_16x16x32_f16(
      /*neg_a=*/false, a, /*neg_b=*/false, b,
      /*c_mod=*/(short)0, c, /*reuse_a=*/false, /*reuse_b=*/false);
}

// A-matrix 16x32 f16 fragment from LDS. Row stride must keep 16B alignment.
// lanes 0-15: M=lane,   halves [kh..kh+7] then [16+kh..16+kh+7], kh=0
// lanes16-31: M=lane-16, same with kh=8   -> two contiguous b128 LDS loads.
__device__ __forceinline__ v16h load_a_lds(const _Float16* base, int stride,
                                           int rowbase, int kbase, int lane) {
  int m  = lane & 15;
  int kh = (lane >> 4) << 3;  // 0 or 8
  const _Float16* row = base + (size_t)(rowbase + m) * stride + kbase + kh;
  v8h lo = *(const v8h*)row;
  v8h hi = *(const v8h*)(row + 16);
  v16h a;
#pragma unroll
  for (int i = 0; i < 8; ++i) { a[i] = lo[i]; a[8 + i] = hi[i]; }
  return a;
}

// B fragment from LDS in pre-swizzled fragment order:
// frag[tile][lane][16 halves] -> two contiguous b128 LDS loads per lane.
__device__ __forceinline__ v16h load_b_frag(const _Float16* fragbase,
                                            int tile, int lane) {
  const _Float16* p16 = fragbase + ((size_t)tile * 32 + lane) * 16;
  v8h lo = *(const v8h*)p16;
  v8h hi = *(const v8h*)(p16 + 8);
  v16h b;
#pragma unroll
  for (int i = 0; i < 8; ++i) { b[i] = lo[i]; b[8 + i] = hi[i]; }
  return b;
}

// ---------------------------------------------------------------------------
// Kernel 0: fold eval-mode BN into weights; emit f16 weights PRE-SWIZZLED in
// WMMA B-fragment order: element h of lane l of tile (ct,kk) is
// B(k = kk*32 + ((l>=16)?8:0) + (h&7) + ((h>=8)?16:0),  n = ct*16 + (l&15)).
// ---------------------------------------------------------------------------
__global__ void prep_kernel(const float* __restrict__ W1, const float* __restrict__ b1,
                            const float* __restrict__ g1, const float* __restrict__ be1,
                            const float* __restrict__ m1, const float* __restrict__ v1,
                            const float* __restrict__ W2, const float* __restrict__ b2,
                            const float* __restrict__ g2, const float* __restrict__ be2,
                            const float* __restrict__ m2, const float* __restrict__ v2,
                            const float* __restrict__ W3,
                            float* __restrict__ W1f, float* __restrict__ b1f,
                            _Float16* __restrict__ W2frag, float* __restrict__ b2f,
                            _Float16* __restrict__ W3frag) {
  int id = blockIdx.x * blockDim.x + threadIdx.x;
  if (id < 192) {                               // W1 folded: 64x3
    int o = id / 3;
    float s = g1[o] * rsqrtf(v1[o] + EPS);
    W1f[id] = W1[id] * s;
  } else if (id < 256) {                        // b1 folded
    int o = id - 192;
    float s = g1[o] * rsqrtf(v1[o] + EPS);
    b1f[o] = b1[o] * s + be1[o] - m1[o] * s;
  } else if (id < 256 + 8192) {                 // W2 folded -> frag order (16 tiles: ct*2+kk)
    int e = id - 256;
    int h = e & 15, lane = (e >> 4) & 31, tile = e >> 9;
    int kk = tile & 1, ct = tile >> 1;
    int k = kk * 32 + ((lane >> 4) << 3) + (h & 7) + ((h >> 3) << 4);
    int o = ct * 16 + (lane & 15);
    float s = g2[o] * rsqrtf(v2[o] + EPS);
    W2frag[e] = (_Float16)(W2[o * 64 + k] * s);
  } else if (id < 256 + 8192 + 128) {           // b2 folded
    int o = id - 256 - 8192;
    float s = g2[o] * rsqrtf(v2[o] + EPS);
    b2f[o] = b2[o] * s + be2[o] - m2[o] * s;
  } else if (id < 256 + 8192 + 128 + 32768) {   // W3 -> frag order (64 tiles: ct*4+kk)
    int e = id - 256 - 8192 - 128;
    int h = e & 15, lane = (e >> 4) & 31, tile = e >> 9;
    int kk = tile & 3, ct = tile >> 2;
    int k = kk * 32 + ((lane >> 4) << 3) + (h & 7) + ((h >> 3) << 4);
    int o = ct * 16 + (lane & 15);
    W3frag[e] = (_Float16)W3[o * 128 + k];
  }
}

// ---------------------------------------------------------------------------
// Kernel 1: furthest point sampling. One block per batch; min_d lives in LDS.
// ---------------------------------------------------------------------------
__global__ void fps_kernel(const float* __restrict__ xyz,
                           float* __restrict__ centers,
                           float* __restrict__ out_centers) {
  extern __shared__ unsigned char smem[];
  float* min_d = (float*)smem;                                   // NPTS
  float* rv    = (float*)(smem + NPTS * sizeof(float));          // 32
  int*   ri    = (int*)(smem + NPTS * sizeof(float) + 32 * 4);   // 32
  __shared__ int s_last;

  const int b   = blockIdx.x;
  const int tid = threadIdx.x;
  const int nt  = blockDim.x;  // 1024 = 32 waves (wave32)
  const float* pts = xyz + (size_t)b * NPTS * 3;

  for (int i = tid; i < NPTS; i += nt) min_d[i] = 1e10f;
  if (tid == 0) {
    float cx = pts[0], cy = pts[1], cz = pts[2];
    float* c = centers + (size_t)b * M * 3;
    c[0] = cx; c[1] = cy; c[2] = cz;
    float* o = out_centers + (size_t)b * M * 3;
    o[0] = cx; o[1] = cy; o[2] = cz;
    s_last = 0;
  }
  __syncthreads();

  int last = 0;
  for (int it = 1; it < M; ++it) {
    float lx = pts[last * 3 + 0], ly = pts[last * 3 + 1], lz = pts[last * 3 + 2];
    float bv = -1.0f;
    int   bi = 0x7fffffff;
    for (int i = tid; i < NPTS; i += nt) {
      float dx = pts[i * 3 + 0] - lx;
      float dy = pts[i * 3 + 1] - ly;
      float dz = pts[i * 3 + 2] - lz;
      float d  = dx * dx + dy * dy + dz * dz;
      float md = fminf(min_d[i], d);
      min_d[i] = md;
      if (md > bv) { bv = md; bi = i; }  // strict > keeps first index (argmax tie rule)
    }
#pragma unroll
    for (int off = 16; off; off >>= 1) {
      float ov = __shfl_xor(bv, off);
      int   oi = __shfl_xor(bi, off);
      if (ov > bv || (ov == bv && oi < bi)) { bv = ov; bi = oi; }
    }
    int wid = tid >> 5, lane = tid & 31;
    if (lane == 0) { rv[wid] = bv; ri[wid] = bi; }
    __syncthreads();
    if (wid == 0) {
      bv = rv[lane];
      bi = ri[lane];
#pragma unroll
      for (int off = 16; off; off >>= 1) {
        float ov = __shfl_xor(bv, off);
        int   oi = __shfl_xor(bi, off);
        if (ov > bv || (ov == bv && oi < bi)) { bv = ov; bi = oi; }
      }
      if (lane == 0) s_last = bi;
    }
    __syncthreads();
    last = s_last;
    if (tid == 0) {
      float cx = pts[last * 3 + 0], cy = pts[last * 3 + 1], cz = pts[last * 3 + 2];
      float* c = centers + ((size_t)b * M + it) * 3;
      c[0] = cx; c[1] = cy; c[2] = cz;
      float* o = out_centers + ((size_t)b * M + it) * 3;
      o[0] = cx; o[1] = cy; o[2] = cz;
    }
  }
}

// ---------------------------------------------------------------------------
// Kernel 2: brute-force KNN (K smallest) + center-relative grouping.
// ---------------------------------------------------------------------------
__global__ void knn_kernel(const float* __restrict__ xyz,
                           const float* __restrict__ centers,
                           float* __restrict__ grouped) {
  extern __shared__ unsigned char smem[];
  float* dist = (float*)smem;                                  // NPTS
  float* rv   = (float*)(smem + NPTS * sizeof(float));         // 8
  int*   ri   = (int*)(smem + NPTS * sizeof(float) + 8 * 4);   // 8
  __shared__ int s_best;

  const int g   = blockIdx.x;           // b*M + m
  const int b   = g / M;
  const int tid = threadIdx.x;
  const int nt  = blockDim.x;           // 256 = 8 waves
  const float* pts = xyz + (size_t)b * NPTS * 3;
  const float cx = centers[g * 3 + 0];
  const float cy = centers[g * 3 + 1];
  const float cz = centers[g * 3 + 2];

  for (int i = tid; i < NPTS; i += nt) {
    float dx = pts[i * 3 + 0] - cx;
    float dy = pts[i * 3 + 1] - cy;
    float dz = pts[i * 3 + 2] - cz;
    dist[i] = dx * dx + dy * dy + dz * dz;
  }
  __syncthreads();

  for (int k = 0; k < K; ++k) {
    float bv = 3.0e38f;
    int   bi = 0x7fffffff;
    for (int i = tid; i < NPTS; i += nt) {
      float d = dist[i];
      if (d < bv) { bv = d; bi = i; }
    }
#pragma unroll
    for (int off = 16; off; off >>= 1) {
      float ov = __shfl_xor(bv, off);
      int   oi = __shfl_xor(bi, off);
      if (ov < bv || (ov == bv && oi < bi)) { bv = ov; bi = oi; }
    }
    int wid = tid >> 5, lane = tid & 31;
    if (lane == 0) { rv[wid] = bv; ri[wid] = bi; }
    __syncthreads();
    if (wid == 0) {
      bv = (lane < 8) ? rv[lane] : 3.0e38f;
      bi = (lane < 8) ? ri[lane] : 0x7fffffff;
#pragma unroll
      for (int off = 4; off; off >>= 1) {
        float ov = __shfl_xor(bv, off);
        int   oi = __shfl_xor(bi, off);
        if (ov < bv || (ov == bv && oi < bi)) { bv = ov; bi = oi; }
      }
      if (lane == 0) s_best = bi;
    }
    __syncthreads();
    int best = s_best;
    if (tid == 0) {
      dist[best] = 3.0e38f;
      float* gr = grouped + ((size_t)g * K + k) * 3;
      gr[0] = pts[best * 3 + 0] - cx;
      gr[1] = pts[best * 3 + 1] - cy;
      gr[2] = pts[best * 3 + 2] - cz;
    }
    __syncthreads();
  }
}

// ---------------------------------------------------------------------------
// Kernel 3: MiniPointNet. One block = 4 waves = 2 centers (64 rows).
// Weights staged to LDS via Tensor Data Mover (tensor_load_to_lds +
// s_wait_tensorcnt), overlapped with scalar layer 1. L2/L3 via
// v_wmma_f32_16x16x32_f16 with all fragments as b128 LDS loads.
// ---------------------------------------------------------------------------
#define H1S 80    // halves; 160B rows -> 16B-aligned fragments
#define H2S 144   // halves; 288B rows -> 16B-aligned fragments
#define OFF_W3   0u
#define OFF_W2   65536u
#define OFF_H1   81920u
#define OFF_H2   92160u
#define OFF_WMAX 110592u
#define OFF_GPTS 114688u
#define SMEM3    115456u

__global__ void mlp_kernel(const float* __restrict__ grouped,
                           const float* __restrict__ centers,
                           const float* __restrict__ W1f, const float* __restrict__ b1f,
                           const _Float16* __restrict__ W2frag, const float* __restrict__ b2f,
                           const _Float16* __restrict__ W3frag, const float* __restrict__ b3,
                           const float* __restrict__ Wp, const float* __restrict__ bp,
                           float* __restrict__ out_tokens) {
  extern __shared__ unsigned char smem3[];
  _Float16* w3s  = (_Float16*)(smem3 + OFF_W3);    // 32768 halves
  _Float16* w2s  = (_Float16*)(smem3 + OFF_W2);    // 8192 halves
  _Float16* h1   = (_Float16*)(smem3 + OFF_H1);    // [64][H1S]
  _Float16* h2   = (_Float16*)(smem3 + OFF_H2);    // [64][H2S]
  float*    wmax = (float*)(smem3 + OFF_WMAX);     // [4][FEAT]
  float*    gpts = (float*)(smem3 + OFF_GPTS);     // [64][3]

  const int tid   = threadIdx.x;      // 128 threads = 4 waves
  const int wave  = tid >> 5;
  const int lane  = tid & 31;
  const int cbase = blockIdx.x * 2;   // two centers per block

  __builtin_prefetch(grouped + (size_t)cbase * K * 3, 0, 1);  // global_prefetch_b8

#if defined(HAVE_TDM)
  // ---- stage weights with the Tensor Data Mover (async DMA to LDS) ----
  if (tid == 0) {
    unsigned dynbase = (unsigned)__builtin_amdgcn_groupstaticsize();
    i32x4 z4 = {0, 0, 0, 0};
#if __clang_major__ >= 23
    i32x8 z8 = {0, 0, 0, 0, 0, 0, 0, 0};
#endif
    // W3frag: 32768 f16 elements, 1-D tile (data_size=2B)
    {
      unsigned long long ga = (unsigned long long)(uintptr_t)W3frag;
      u32x4 g0;
      g0[0] = 1u;                                   // count=1 (valid descriptor)
      g0[1] = dynbase + OFF_W3;                     // lds_addr
      g0[2] = (unsigned)ga;                         // global_addr[31:0]
      g0[3] = (unsigned)((ga >> 32) & 0x01FFFFFFu) | (2u << 30);  // addr[56:32] | type=2
      i32x8 g1 = {0x00010000,        // wg_mask=0, data_size=1 (2B)
                  (int)0x80000000u,  // tensor_dim0 = 32768 (low16 in [31:16])
                  0x00010000,        // dim0 hi=0, tensor_dim1 = 1
                  (int)0x80000000u,  // dim1 hi=0, tile_dim0 = 32768
                  0,                 // tile_dim1=0, tile_dim2=0
                  32768,             // tensor_dim0_stride low32
                  0, 0};
#if __clang_major__ >= 23
      __builtin_amdgcn_tensor_load_to_lds(g0, g1, z4, z4, z8, 0);
#else
      __builtin_amdgcn_tensor_load_to_lds(g0, g1, z4, z4, 0);
#endif
    }
    // W2frag: 8192 f16 elements
    {
      unsigned long long ga = (unsigned long long)(uintptr_t)W2frag;
      u32x4 g0;
      g0[0] = 1u;
      g0[1] = dynbase + OFF_W2;
      g0[2] = (unsigned)ga;
      g0[3] = (unsigned)((ga >> 32) & 0x01FFFFFFu) | (2u << 30);
      i32x8 g1 = {0x00010000,
                  0x20000000,        // tensor_dim0 = 8192
                  0x00010000,        // tensor_dim1 = 1
                  0x20000000,        // tile_dim0 = 8192
                  0,
                  8192,
                  0, 0};
#if __clang_major__ >= 23
      __builtin_amdgcn_tensor_load_to_lds(g0, g1, z4, z4, z8, 0);
#else
      __builtin_amdgcn_tensor_load_to_lds(g0, g1, z4, z4, 0);
#endif
    }
  }
#else
  // ---- fallback: cooperative b128 copy of both weight arrays ----
  {
    const uint4* s3 = (const uint4*)W3frag;
    uint4*       d3 = (uint4*)w3s;
    for (int i = tid; i < 4096; i += 128) d3[i] = s3[i];
    const uint4* s2 = (const uint4*)W2frag;
    uint4*       d2 = (uint4*)w2s;
    for (int i = tid; i < 1024; i += 128) d2[i] = s2[i];
  }
#endif

  // stage grouped points (2 centers x 32 nbrs x 3) — overlaps with TDM
  for (int e = tid; e < 192; e += 128)
    gpts[e] = grouped[(size_t)cbase * K * 3 + e];
  __syncthreads();

  // layer 1: 3 -> 64, BN+ReLU folded, f16 into LDS (overlaps with TDM)
  for (int e = tid; e < 4096; e += 128) {
    int r = e >> 6, o = e & 63;
    float x = gpts[r * 3 + 0] * W1f[o * 3 + 0] +
              gpts[r * 3 + 1] * W1f[o * 3 + 1] +
              gpts[r * 3 + 2] * W1f[o * 3 + 2] + b1f[o];
    h1[r * H1S + o] = (_Float16)fmaxf(x, 0.0f);
  }
#if defined(HAVE_TDM)
  if (tid == 0) {
#if __has_builtin(__builtin_amdgcn_s_wait_tensorcnt)
    __builtin_amdgcn_s_wait_tensorcnt(0);
#else
    asm volatile("s_wait_tensorcnt 0x0" ::: "memory");
#endif
  }
#endif
  __syncthreads();

  const int rowbase = wave * 16;

  // layer 2: 64 -> 128 via WMMA (8 col tiles x 2 k-steps), weights from LDS
  for (int ct = 0; ct < 8; ++ct) {
    v8f acc;
#pragma unroll
    for (int r = 0; r < 8; ++r) acc[r] = 0.0f;
#pragma unroll
    for (int kk = 0; kk < 2; ++kk) {
      v16h a  = load_a_lds(h1, H1S, rowbase, kk * 32, lane);
      v16h bb = load_b_frag(w2s, ct * 2 + kk, lane);
      acc = wmma16(a, bb, acc);
    }
    int n = lane & 15, mo = (lane >> 4) << 3;
    float bias = b2f[ct * 16 + n];
#pragma unroll
    for (int r = 0; r < 8; ++r) {
      float x = acc[r] + bias;
      h2[(rowbase + mo + r) * H2S + ct * 16 + n] = (_Float16)fmaxf(x, 0.0f);
    }
  }
  __syncthreads();

  // layer 3: 128 -> 256 via WMMA (16 col tiles x 4 k-steps) + row-max
  for (int ct = 0; ct < 16; ++ct) {
    v8f acc;
#pragma unroll
    for (int r = 0; r < 8; ++r) acc[r] = 0.0f;
#pragma unroll
    for (int kk = 0; kk < 4; ++kk) {
      v16h a  = load_a_lds(h2, H2S, rowbase, kk * 32, lane);
      v16h bb = load_b_frag(w3s, ct * 4 + kk, lane);
      acc = wmma16(a, bb, acc);
    }
    int n = lane & 15;
    float bias = b3[ct * 16 + n];
    float mx = -3.0e38f;
#pragma unroll
    for (int r = 0; r < 8; ++r) mx = fmaxf(mx, acc[r] + bias);
    // lanes l and l+16 hold rows 0-7 / 8-15 of the same column -> combine
    float om = __shfl_xor(mx, 16);
    mx = fmaxf(mx, om);
    if (lane < 16) wmax[wave * FEAT + ct * 16 + n] = mx;
  }
  __syncthreads();

  // combine the 2 waves per center (32 rows total) + positional embed
  for (int e = tid; e < 2 * FEAT; e += 128) {
    int c = e >> 8, col = e & 255;
    int g = cbase + c;
    float t  = fmaxf(wmax[2 * c * FEAT + col], wmax[(2 * c + 1) * FEAT + col]);
    float px = centers[g * 3 + 0], py = centers[g * 3 + 1], pz = centers[g * 3 + 2];
    float pos = Wp[col * 3 + 0] * px + Wp[col * 3 + 1] * py +
                Wp[col * 3 + 2] * pz + bp[col];
    out_tokens[(size_t)g * FEAT + col] = t + pos;
  }
}

// ---------------------------------------------------------------------------
extern "C" void kernel_launch(void* const* d_in, const int* in_sizes, int n_in,
                              void* d_out, int out_size, void* d_ws, size_t ws_size,
                              hipStream_t stream) {
  (void)in_sizes; (void)n_in; (void)out_size; (void)ws_size;
  const float* xyz = (const float*)d_in[0];
  const float* W1  = (const float*)d_in[1];
  const float* b1  = (const float*)d_in[2];
  const float* g1  = (const float*)d_in[3];
  const float* be1 = (const float*)d_in[4];
  const float* m1  = (const float*)d_in[5];
  const float* v1  = (const float*)d_in[6];
  const float* W2  = (const float*)d_in[7];
  const float* b2  = (const float*)d_in[8];
  const float* g2  = (const float*)d_in[9];
  const float* be2 = (const float*)d_in[10];
  const float* m2  = (const float*)d_in[11];
  const float* v2  = (const float*)d_in[12];
  const float* W3  = (const float*)d_in[13];
  const float* b3  = (const float*)d_in[14];
  const float* Wp  = (const float*)d_in[15];
  const float* bp  = (const float*)d_in[16];

  float* out_centers = (float*)d_out;                          // (B,M,3)
  float* out_tokens  = (float*)d_out + (size_t)BATCH * M * 3;  // (B,M,FEAT)

  // workspace carve (256B aligned)
  char* wsb = (char*)d_ws;
  size_t off = 0;
  auto alloc = [&](size_t bytes) -> char* {
    char* p = wsb + off;
    off = (off + bytes + 255) & ~(size_t)255;
    return p;
  };
  float*     centers = (float*)alloc((size_t)BATCH * M * 3 * sizeof(float));
  float*     grouped = (float*)alloc((size_t)BATCH * M * K * 3 * sizeof(float));
  float*     W1f     = (float*)alloc(192 * sizeof(float));
  float*     b1f     = (float*)alloc(64 * sizeof(float));
  _Float16*  W2frag  = (_Float16*)alloc(64 * 128 * sizeof(_Float16));
  float*     b2f     = (float*)alloc(128 * sizeof(float));
  _Float16*  W3frag  = (_Float16*)alloc(128 * 256 * sizeof(_Float16));

  // 0) fold BN + convert/swizzle weights into WMMA fragment order
  prep_kernel<<<162, 256, 0, stream>>>(W1, b1, g1, be1, m1, v1,
                                       W2, b2, g2, be2, m2, v2, W3,
                                       W1f, b1f, W2frag, b2f, W3frag);
  // 1) FPS (min_d in 64KB dynamic LDS)
  size_t smem1 = (size_t)NPTS * 4 + 32 * 4 + 32 * 4;
  fps_kernel<<<BATCH, 1024, smem1, stream>>>(xyz, centers, out_centers);
  // 2) KNN + grouping
  size_t smem2 = (size_t)NPTS * 4 + 8 * 4 + 8 * 4;
  knn_kernel<<<BATCH * M, 256, smem2, stream>>>(xyz, centers, grouped);
  // 3) MiniPointNet with WMMA + TDM-staged weights
  mlp_kernel<<<BATCH * M / 2, 128, SMEM3, stream>>>(grouped, centers,
                                                    W1f, b1f, W2frag, b2f,
                                                    W3frag, b3, Wp, bp,
                                                    out_tokens);
}
